// ParticleNet_27943057227833
// MI455X (gfx1250) — matrix operand order repaired and checked by reading, not compiled, not run on hardware
//
#include <hip/hip_runtime.h>
#include <hip/hip_bf16.h>

// ---------------------------------------------------------------------------
// ParticleNet forward for MI455X (gfx1250, wave32, WMMA).
// All GEMMs (edge convs, fusion, FCs) use v_wmma_f32_16x16x32_f16 (f16 A/B,
// f32 accumulate). kNN top-K stays fp32 (ordering-sensitive).
// GEMM waves process 2 column tiles sharing one A fragment: halves A traffic
// and provides two independent WMMA accumulate chains for latency hiding.
// Inner k-loops are branch-free so the compiler can clause/pipeline loads.
// ---------------------------------------------------------------------------

#define NB   256   // events
#define NP   128   // points per event
#define KNN  16    // neighbors
#define CSTR 256   // channel stride of feature buffers (max channels)
#define FUSED 448  // 64 + 128 + 256 concat channels

typedef __attribute__((ext_vector_type(16))) _Float16 v16h;
typedef __attribute__((ext_vector_type(8)))  _Float16 v8h;
typedef __attribute__((ext_vector_type(8)))  float    v8f;

// ---------------------------------------------------------------------------
// WMMA fragment helpers (A/B k order is this code's contiguous choice, not the ISA table order)
// A 16x32 f16: lane m = lane&15, khalf = lane>>4;
//   halves 0..15 -> K = 16*khalf + 0..15
// B 32x16 f16: lane n = lane&15, khalf = lane>>4; halves e -> K = 16*khalf + e
//   (pre-packed so each lane's 16 halves are 32 contiguous bytes)
// D f32: vgpr v, lane l -> M = v + 8*(l>>4), N = l&15
// ---------------------------------------------------------------------------
__device__ __forceinline__ v16h load_a_frag(const _Float16* tile, int rowStride,
                                            int k0, int lane) {
  int m  = lane & 15;
  int kh = (lane >> 4) & 1;
  const _Float16* p = tile + (size_t)m * rowStride + k0 + kh * 16;
  v8h lo = *(const v8h*)p;         // K = k0+16*kh .. +7   (16B aligned)
  v8h hi = *(const v8h*)(p + 8);   // K = k0+16*kh+8 .. +7
  v16h a;
#pragma unroll
  for (int i = 0; i < 8; ++i) { a[i] = lo[i]; a[8 + i] = hi[i]; }
  return a;
}

__device__ __forceinline__ v16h load_b_frag(const _Float16* wp, int fragIdx,
                                            int lane) {
  const _Float16* p = wp + ((size_t)fragIdx * 32 + lane) * 16;  // 32B/lane
  v8h lo = *(const v8h*)p;
  v8h hi = *(const v8h*)(p + 8);
  v16h b;
#pragma unroll
  for (int i = 0; i < 8; ++i) { b[i] = lo[i]; b[8 + i] = hi[i]; }
  return b;
}

// ---------------------------------------------------------------------------
// Pack fp32 weight matrix W[Co][Ci] (row-major) into B-fragment order fp16,
// zero-padded to Kpad (mult 32) x CoPad (mult 16).
// element index: tid = ((kt*nCot + cot)*32 + lane)*16 + e
// ---------------------------------------------------------------------------
__global__ void pack_w_kernel(const float* __restrict__ W, _Float16* __restrict__ out,
                              int Co, int Ci, int CoPad, int Kpad) {
  int tid = blockIdx.x * blockDim.x + threadIdx.x;
  int total = Kpad * CoPad;
  if (tid >= total) return;
  int e    = tid & 15;
  int lane = (tid >> 4) & 31;
  int f    = tid >> 9;
  int nCot = CoPad >> 4;
  int cot  = f % nCot;
  int kt   = f / nCot;
  int K  = kt * 32 + ((lane >> 4) & 1) * 16 + e;
  int co = cot * 16 + (lane & 15);
  float v = (co < Co && K < Ci) ? W[(size_t)co * Ci + K] : 0.0f;
  out[tid] = (_Float16)v;
}

// ---------------------------------------------------------------------------
// BatchNorm on input features -> fp32 + fp16 feature buffers (CSTR stride,
// zero padded so downstream vector loads read defined data).
// ---------------------------------------------------------------------------
__global__ void bn_feats_kernel(const float* __restrict__ features,
                                const float* __restrict__ mask,
                                const float* __restrict__ scale,
                                const float* __restrict__ bias,
                                float* __restrict__ f32, _Float16* __restrict__ f16) {
  int tid = blockIdx.x * blockDim.x + threadIdx.x;
  int c  = tid % CSTR;
  int np = tid / CSTR;
  if (np >= NB * NP) return;
  float v = 0.0f;
  if (c < 7) v = (features[(size_t)np * 7 + c] * scale[c] + bias[c]) * mask[np];
  f32[(size_t)np * CSTR + c] = v;
  f16[(size_t)np * CSTR + c] = (_Float16)v;
}

// ---------------------------------------------------------------------------
// kNN: one block per event, 128 threads (one per point). Points staged in LDS
// (with coord_shift), per-thread insertion sort of K+1 smallest distances.
// Strict '<' keeps the earlier index on ties, matching lax.top_k ordering.
// ---------------------------------------------------------------------------
template <int CP>
__global__ void knn_kernel(const float* __restrict__ pts, int stride,
                           const float* __restrict__ mask, int* __restrict__ idx) {
  __shared__ float sp[NP * CP];
  int n = blockIdx.x;
  int p = threadIdx.x;
  float shift = (mask[n * NP + p] == 0.0f) ? 1e9f : 0.0f;
  for (int c = 0; c < CP; ++c)
    sp[p * CP + c] = pts[((size_t)n * NP + p) * stride + c] + shift;
  __syncthreads();

  float bd[KNN + 1];
  int   bi[KNN + 1];
#pragma unroll
  for (int j = 0; j <= KNN; ++j) { bd[j] = 3.4e38f; bi[j] = 0; }
  for (int q = 0; q < NP; ++q) {
    float d = 0.0f;
    for (int c = 0; c < CP; ++c) {
      float t = sp[p * CP + c] - sp[q * CP + c];
      d += t * t;
    }
    if (d < bd[KNN]) {
      int j = KNN;
      while (j > 0 && d < bd[j - 1]) { bd[j] = bd[j - 1]; bi[j] = bi[j - 1]; --j; }
      bd[j] = d; bi[j] = q;
    }
  }
  for (int k = 0; k < KNN; ++k)
    idx[((size_t)n * NP + p) * KNN + k] = bi[k + 1];  // drop self (entry 0)
}

// ---------------------------------------------------------------------------
// One WMMA GEMM layer over a 256-row tile living in LDS.
// Row tiles (16) are partitioned across the 8 waves; each wave reads and
// writes only its own rows, so no intra-layer barrier is needed.
// Column tiles processed in pairs: one A fragment feeds two WMMA chains.
// nCt is even for all instantiations (CO = 64/128/256). Branch-free body.
// ---------------------------------------------------------------------------
template <int CO>
__device__ __forceinline__ void gemm_layer(const _Float16* Asrc, int Kpad,
                                           const _Float16* __restrict__ Wp,
                                           const float* __restrict__ sc,
                                           const float* __restrict__ bi,
                                           _Float16* Ydst, int lane, int wave) {
  const int nCt = CO / 16;
  for (int rt = wave; rt < 16; rt += 8) {
    const _Float16* atile = Asrc + (size_t)rt * 16 * Kpad;
    for (int ct = 0; ct < nCt; ct += 2) {
      v8f acc0 = {0.f, 0.f, 0.f, 0.f, 0.f, 0.f, 0.f, 0.f};
      v8f acc1 = {0.f, 0.f, 0.f, 0.f, 0.f, 0.f, 0.f, 0.f};
      for (int kt = 0; kt < Kpad / 32; ++kt) {
        v16h a  = load_a_frag(atile, Kpad, kt * 32, lane);
        v16h b0 = load_b_frag(Wp, kt * nCt + ct, lane);
        v16h b1 = load_b_frag(Wp, kt * nCt + ct + 1, lane);
        acc0 = __builtin_amdgcn_wmma_f32_16x16x32_f16(
            false, a, false, b0, (short)0, acc0, false, false);
        acc1 = __builtin_amdgcn_wmma_f32_16x16x32_f16(
            false, a, false, b1, (short)0, acc1, false, false);
      }
      int co0 = ct * 16 + (lane & 15);
      int co1 = co0 + 16;
      float s0 = sc[co0], bb0 = bi[co0];
      float s1 = sc[co1], bb1 = bi[co1];
      int mbase = rt * 16 + 8 * ((lane >> 4) & 1);
#pragma unroll
      for (int v = 0; v < 8; ++v) {
        float y0 = acc0[v] * s0 + bb0;
        float y1 = acc1[v] * s1 + bb1;
        y0 = y0 > 0.0f ? y0 : 0.0f;              // relu
        y1 = y1 > 0.0f ? y1 : 0.0f;
        Ydst[(size_t)(mbase + v) * CO + co0] = (_Float16)y0;
        Ydst[(size_t)(mbase + v) * CO + co1] = (_Float16)y1;
      }
    }
  }
}

// ---------------------------------------------------------------------------
// EdgeConv block: one workgroup = one event x 16 points (256 edge rows).
// Layer1 A = [center, nbr-center] staged into LDS (gather hits L2: the f16
// feature tensor is <=17MB, L2 is 192MB). Three GEMM layers ping-pong in LDS,
// then mean-over-k + fp32 shortcut epilogue.
// Dynamic LDS = 2 * 256*CO*2 bytes (<=256KB of the 320KB WGP budget).
// ---------------------------------------------------------------------------
template <int CIN, int CO>
__global__ void edge_block_kernel(
    const _Float16* __restrict__ fin16, const float* __restrict__ fin32,
    const int* __restrict__ idx,
    const _Float16* __restrict__ w1, const float* __restrict__ s1, const float* __restrict__ b1,
    const _Float16* __restrict__ w2, const float* __restrict__ s2, const float* __restrict__ b2,
    const _Float16* __restrict__ w3, const float* __restrict__ s3, const float* __restrict__ b3,
    const float* __restrict__ scw, const float* __restrict__ scs, const float* __restrict__ scb,
    const float* __restrict__ mask,
    float* __restrict__ fout32, _Float16* __restrict__ fout16,
    _Float16* __restrict__ fused, int fusedOff) {
  constexpr int K1 = ((2 * CIN) + 31) / 32 * 32;  // layer-1 K (zero padded)
  extern __shared__ _Float16 smem[];
  _Float16* Y0 = smem;                       // 256 * CO halves
  _Float16* R1 = smem + (size_t)256 * CO;    // A-slab (256*K1) then Y1 (256*CO)

  int n  = blockIdx.x;
  int p0 = blockIdx.y * 16;
  int lane = threadIdx.x & 31;
  int wave = threadIdx.x >> 5;

  // ---- stage layer-1 edge features: row r = (point, neighbor) ----
  {
    int r = threadIdx.x;                     // 256 threads == 256 rows
    int p = p0 + (r >> 4);
    int k = r & 15;
    int nb = idx[((size_t)n * NP + p) * KNN + k];
    const _Float16* fc = fin16 + ((size_t)n * NP + p)  * CSTR;
    const _Float16* fn = fin16 + ((size_t)n * NP + nb) * CSTR;
    _Float16* row = R1 + (size_t)r * K1;
    for (int c = 0; c < K1; ++c) {
      _Float16 v = (_Float16)0.0f;
      if (c < CIN)            v = fc[c];
      else if (c < 2 * CIN)   v = (_Float16)(fn[c - CIN] - fc[c - CIN]);
      row[c] = v;
    }
  }
  __syncthreads();

  gemm_layer<CO>(R1, K1, w1, s1, b1, Y0, lane, wave);  // L1: slab -> Y0
  __syncthreads();
  gemm_layer<CO>(Y0, CO, w2, s2, b2, R1, lane, wave);  // L2: Y0 -> Y1
  __syncthreads();
  gemm_layer<CO>(R1, CO, w3, s3, b3, Y0, lane, wave);  // L3: Y1 -> Y0
  __syncthreads();

  // ---- epilogue: mean over K, fp32 shortcut, residual, mask ----
  for (int o = threadIdx.x; o < 16 * CO; o += 256) {
    int pi = o / CO;
    int co = o % CO;
    float m = 0.0f;
    for (int k = 0; k < KNN; ++k) m += (float)Y0[(size_t)(pi * 16 + k) * CO + co];
    m *= (1.0f / KNN);
    const float* f = fin32 + ((size_t)n * NP + p0 + pi) * CSTR;
    const float* w = scw + (size_t)co * CIN;
    float d = 0.0f;
    for (int c = 0; c < CIN; ++c) d += f[c] * w[c];
    d = d * scs[co] + scb[co];
    d = d > 0.0f ? d : 0.0f;
    float out = (m + d) * mask[n * NP + p0 + pi];
    size_t base = (size_t)n * NP + p0 + pi;
    fout32[base * CSTR + co]            = out;
    fout16[base * CSTR + co]            = (_Float16)out;
    fused[base * FUSED + fusedOff + co] = (_Float16)out;
  }
}

// ---------------------------------------------------------------------------
// Generic WMMA GEMM: out[m,co] = act(A[m,:] . W[co,:] * scale + bias).
// One wave per 16x32 output strip (two 16x16 tiles sharing the A fragment).
// A row-major fp16 in global (row stride = Kpad halves, 16B-aligned),
// W pre-packed fragments. M must be mult of 16. The single-tile tail case is
// hoisted OUT of the k-loop so both hot loops are branch-free and the
// scheduler can clause + pipeline the global_load_b128 stream.
// ---------------------------------------------------------------------------
__global__ void wmma_gemm_kernel(const _Float16* __restrict__ A,
                                 const _Float16* __restrict__ Wp,
                                 const float* __restrict__ scale,
                                 const float* __restrict__ bias,
                                 int Kpad, int CoPad, int CoReal, int relu,
                                 float* __restrict__ out32,
                                 _Float16* __restrict__ out16) {
  int rt  = blockIdx.x;
  int ct0 = blockIdx.y * 2;
  int lane = threadIdx.x & 31;
  int nCt = CoPad >> 4;
  bool two = (ct0 + 1) < nCt;                    // uniform across the wave
  const _Float16* atile = A + (size_t)rt * 16 * Kpad;
  v8f acc0 = {0.f, 0.f, 0.f, 0.f, 0.f, 0.f, 0.f, 0.f};
  v8f acc1 = {0.f, 0.f, 0.f, 0.f, 0.f, 0.f, 0.f, 0.f};
  if (two) {
    for (int kt = 0; kt < (Kpad >> 5); ++kt) {
      v16h a  = load_a_frag(atile, Kpad, kt * 32, lane);
      v16h b0 = load_b_frag(Wp, kt * nCt + ct0, lane);
      v16h b1 = load_b_frag(Wp, kt * nCt + ct0 + 1, lane);
      acc0 = __builtin_amdgcn_wmma_f32_16x16x32_f16(
          false, a, false, b0, (short)0, acc0, false, false);
      acc1 = __builtin_amdgcn_wmma_f32_16x16x32_f16(
          false, a, false, b1, (short)0, acc1, false, false);
    }
  } else {
    for (int kt = 0; kt < (Kpad >> 5); ++kt) {
      v16h a  = load_a_frag(atile, Kpad, kt * 32, lane);
      v16h b0 = load_b_frag(Wp, kt * nCt + ct0, lane);
      acc0 = __builtin_amdgcn_wmma_f32_16x16x32_f16(
          false, a, false, b0, (short)0, acc0, false, false);
    }
  }
  int mbase = rt * 16 + 8 * ((lane >> 4) & 1);
  int nt = two ? 2 : 1;
  for (int t = 0; t < nt; ++t) {
    v8f acc = t ? acc1 : acc0;
    int co = (ct0 + t) * 16 + (lane & 15);
    float s  = (scale != nullptr && co < CoReal) ? scale[co] : 1.0f;
    float bb = (bias  != nullptr && co < CoReal) ? bias[co]  : 0.0f;
#pragma unroll
    for (int v = 0; v < 8; ++v) {
      float y = acc[v] * s + bb;
      if (relu) y = y > 0.0f ? y : 0.0f;
      if (co >= CoReal) y = 0.0f;
      size_t o = (size_t)(mbase + v) * CoPad + co;
      if (out32) out32[o] = y;
      if (out16) out16[o] = (_Float16)y;
    }
  }
}

// ---------------------------------------------------------------------------
// Masked mean pool over points: pooled[n,co] = sum_p f[n,p,co]*mask / counts.
// ---------------------------------------------------------------------------
__global__ void pool_kernel(const float* __restrict__ f,
                            const float* __restrict__ mask,
                            float* __restrict__ pooled32,
                            _Float16* __restrict__ pooled16) {
  __shared__ float cnt;
  int n = blockIdx.x;
  if (threadIdx.x == 0) {
    float c = 0.0f;
    for (int p = 0; p < NP; ++p) c += mask[n * NP + p];
    cnt = c;
  }
  __syncthreads();
  float inv = 1.0f / cnt;
  for (int co = threadIdx.x; co < 384; co += blockDim.x) {
    float s = 0.0f;
    for (int p = 0; p < NP; ++p)
      s += f[((size_t)n * NP + p) * 384 + co] * mask[n * NP + p];
    s *= inv;
    pooled32[(size_t)n * 384 + co] = s;
    pooled16[(size_t)n * 384 + co] = (_Float16)s;
  }
}

__global__ void copy_out_kernel(const float* __restrict__ src, float* __restrict__ dst) {
  int t = blockIdx.x * blockDim.x + threadIdx.x;
  if (t < NB * 10) dst[t] = src[(t / 10) * 16 + (t % 10)];
}

// ---------------------------------------------------------------------------
// Host orchestration.
// Input order = JAX tree-flatten of setup_inputs() dict (dict keys sorted):
//   0 features, 1 mask, 2 bn_fts_bias, 3 bn_fts_scale,
//   4.. edge[b] (b=0..2, 12 each): convs[l]{bias,scale,w} l=0..2, sc_bias,
//       sc_scale, sc_w
//   40 fc_b, 41 fc_final_b, 42 fc_final_w, 43 fc_w,
//   44 fusion_bias, 45 fusion_scale, 46 fusion_w, 47 points
// ---------------------------------------------------------------------------
extern "C" void kernel_launch(void* const* d_in, const int* in_sizes, int n_in,
                              void* d_out, int out_size, void* d_ws, size_t ws_size,
                              hipStream_t stream) {
  (void)in_sizes; (void)n_in; (void)out_size; (void)ws_size;
  auto F = [&](int i) { return (const float*)d_in[i]; };
  const int IN_FEATURES = 0, IN_MASK = 1, IN_BN_BIAS = 2, IN_BN_SCALE = 3;
  const int E0 = 4, EBLK = 12;
  const int IN_FC_B = 40, IN_FCF_B = 41, IN_FCF_W = 42, IN_FC_W = 43;
  const int IN_FUS_B = 44, IN_FUS_S = 45, IN_FUS_W = 46, IN_POINTS = 47;

  char* ws = (char*)d_ws;
  size_t off = 0;
  auto alloc = [&](size_t bytes) -> char* {
    char* p = ws + off;
    off = (off + bytes + 255) & ~(size_t)255;
    return p;
  };
  float*     fA32 = (float*)alloc((size_t)NB * NP * CSTR * 4);
  float*     fB32 = (float*)alloc((size_t)NB * NP * CSTR * 4);
  _Float16*  fA16 = (_Float16*)alloc((size_t)NB * NP * CSTR * 2);
  _Float16*  fB16 = (_Float16*)alloc((size_t)NB * NP * CSTR * 2);
  int*       idx  = (int*)alloc((size_t)NB * NP * KNN * 4);
  _Float16*  fused = (_Float16*)alloc((size_t)NB * NP * FUSED * 2);
  float*     fuso32  = (float*)alloc((size_t)NB * NP * 384 * 4);
  float*     pooled32 = (float*)alloc((size_t)NB * 384 * 4);
  _Float16*  pooled16 = (_Float16*)alloc((size_t)NB * 384 * 2);
  float*     h32  = (float*)alloc((size_t)NB * 256 * 4);
  _Float16*  h16  = (_Float16*)alloc((size_t)NB * 256 * 2);
  float*     fc2o = (float*)alloc((size_t)NB * 16 * 4);

  auto packW = [&](const float* W, int Co, int Ci, int CoPad, int Kpad) -> _Float16* {
    _Float16* dst = (_Float16*)alloc((size_t)Kpad * CoPad * 2);
    int tot = Kpad * CoPad;
    pack_w_kernel<<<(tot + 255) / 256, 256, 0, stream>>>(W, dst, Co, Ci, CoPad, Kpad);
    return dst;
  };

  const int cin[3] = {7, 64, 128};
  const int cco[3] = {64, 128, 256};
  _Float16* wp[3][3];
  for (int b = 0; b < 3; ++b)
    for (int l = 0; l < 3; ++l) {
      int Ci = (l == 0) ? 2 * cin[b] : cco[b];
      int Kp = (Ci + 31) / 32 * 32;
      wp[b][l] = packW(F(E0 + EBLK * b + 3 * l + 2), cco[b], Ci, cco[b], Kp);
    }
  _Float16* wfus = packW(F(IN_FUS_W), 384, FUSED, 384, FUSED);
  _Float16* wfc1 = packW(F(IN_FC_W),  256, 384, 256, 384);
  _Float16* wfc2 = packW(F(IN_FCF_W), 10,  256, 16,  256);

  // BatchNorm of input features
  bn_feats_kernel<<<(NB * NP * CSTR) / 256, 256, 0, stream>>>(
      F(IN_FEATURES), F(IN_MASK), F(IN_BN_SCALE), F(IN_BN_BIAS), fA32, fA16);

  // ---- EdgeConv block 0 (pts = raw points, C=2; feats 7 -> 64) ----
  knn_kernel<2><<<NB, NP, 0, stream>>>(F(IN_POINTS), 2, F(IN_MASK), idx);
  {
    int b = 0;
    edge_block_kernel<7, 64><<<dim3(NB, NP / 16), 256, 2 * 256 * 64 * 2, stream>>>(
        fA16, fA32, idx,
        wp[b][0], F(E0 + EBLK * b + 1), F(E0 + EBLK * b + 0),
        wp[b][1], F(E0 + EBLK * b + 4), F(E0 + EBLK * b + 3),
        wp[b][2], F(E0 + EBLK * b + 7), F(E0 + EBLK * b + 6),
        F(E0 + EBLK * b + 11), F(E0 + EBLK * b + 10), F(E0 + EBLK * b + 9),
        F(IN_MASK), fB32, fB16, fused, 0);
  }

  // ---- EdgeConv block 1 (pts = feats C=64; feats 64 -> 128) ----
  knn_kernel<64><<<NB, NP, 0, stream>>>(fB32, CSTR, F(IN_MASK), idx);
  {
    int b = 1;
    edge_block_kernel<64, 128><<<dim3(NB, NP / 16), 256, 2 * 256 * 128 * 2, stream>>>(
        fB16, fB32, idx,
        wp[b][0], F(E0 + EBLK * b + 1), F(E0 + EBLK * b + 0),
        wp[b][1], F(E0 + EBLK * b + 4), F(E0 + EBLK * b + 3),
        wp[b][2], F(E0 + EBLK * b + 7), F(E0 + EBLK * b + 6),
        F(E0 + EBLK * b + 11), F(E0 + EBLK * b + 10), F(E0 + EBLK * b + 9),
        F(IN_MASK), fA32, fA16, fused, 64);
  }

  // ---- EdgeConv block 2 (pts = feats C=128; feats 128 -> 256) ----
  knn_kernel<128><<<NB, NP, 0, stream>>>(fA32, CSTR, F(IN_MASK), idx);
  {
    int b = 2;
    edge_block_kernel<128, 256><<<dim3(NB, NP / 16), 256, 2 * 256 * 256 * 2, stream>>>(
        fA16, fA32, idx,
        wp[b][0], F(E0 + EBLK * b + 1), F(E0 + EBLK * b + 0),
        wp[b][1], F(E0 + EBLK * b + 4), F(E0 + EBLK * b + 3),
        wp[b][2], F(E0 + EBLK * b + 7), F(E0 + EBLK * b + 6),
        F(E0 + EBLK * b + 11), F(E0 + EBLK * b + 10), F(E0 + EBLK * b + 9),
        F(IN_MASK), fB32, fB16, fused, 192);
  }

  // ---- Fusion GEMM: (NB*NP) x 448 -> 384, relu(scale,bias) ----
  wmma_gemm_kernel<<<dim3((NB * NP) / 16, 384 / 32), 32, 0, stream>>>(
      fused, wfus, F(IN_FUS_S), F(IN_FUS_B), FUSED, 384, 384, 1, fuso32, nullptr);

  // ---- masked mean pool over points ----
  pool_kernel<<<NB, 256, 0, stream>>>(fuso32, F(IN_MASK), pooled32, pooled16);

  // ---- fc1: 256 x 384 -> 256, relu(+bias) ----
  wmma_gemm_kernel<<<dim3(NB / 16, 256 / 32), 32, 0, stream>>>(
      pooled16, wfc1, nullptr, F(IN_FC_B), 384, 256, 256, 1, h32, h16);

  // ---- fc_final: 256 x 256 -> 10 (padded to 16, single col tile) ----
  wmma_gemm_kernel<<<dim3(NB / 16, 1), 32, 0, stream>>>(
      h16, wfc2, nullptr, F(IN_FCF_B), 256, 16, 10, 0, fc2o, nullptr);

  copy_out_kernel<<<(NB * 10 + 255) / 256, 256, 0, stream>>>(fc2o, (float*)d_out);
}
